// Mlp_cnn_shift_80865644249612
// MI455X (gfx1250) — compile-verified, matrix operand order checked by reading, not run on hardware
//
#include <hip/hip_runtime.h>

typedef __attribute__((ext_vector_type(16))) __bf16 v16bf;
typedef __attribute__((ext_vector_type(8)))  float  v8f;

#define TM 256
#define TN 128
#define TK 32
#define LDP 40   // LDS row stride in ushorts: 80 bytes = 16B-aligned rows, 20-dword bank stride

// SHIFTS = [(1,1),(1,0),(1,-1),(0,1),(0,0),(0,-1),(-1,1),(-1,0),(-1,-1)]
__constant__ int c_sh[9] = { 1, 1, 1, 0, 0, 0, -1, -1, -1 };
__constant__ int c_sw[9] = { 1, 0, -1, 1, 0, -1, 1, 0, -1 };

__device__ __forceinline__ unsigned short f2bf(float f) {
    union { float f; unsigned u; } v; v.f = f;
    unsigned r = v.u + 0x7FFFu + ((v.u >> 16) & 1u);   // round-to-nearest-even
    return (unsigned short)(r >> 16);
}
__device__ __forceinline__ unsigned pack2bf(float a, float b) {
    return (unsigned)f2bf(a) | ((unsigned)f2bf(b) << 16);
}
__device__ __forceinline__ float bf2f(unsigned short h) {
    union { unsigned u; float f; } v; v.u = ((unsigned)h) << 16;
    return v.f;
}
__device__ __forceinline__ float gelu_f(float v) {
    return 0.5f * v * (1.0f + erff(v * 0.70710678118654752f));
}

union AF { v16bf v; uint4 q[2]; unsigned u[8]; };

// MODE 0: A = f32 x,        out = gelu -> bf16 (xh)
// MODE 1: A = fwd-shift gather of bf16 xh (groups of 114), out = gelu -> bf16 (hpre)
// MODE 2: A = bf16 xh,      out = gelu -> bf16 (wbuf)
// MODE 3: A = a0*hgather(inv-shift, groups of 57) + a1*w -> bf16, out = f32 (final)
template<int MODE>
__global__ __launch_bounds__(256)
void gemm_tile(const float* __restrict__ Af32,
               const unsigned short* __restrict__ Abf,
               const float* __restrict__ Bw,
               const float* __restrict__ bias,
               unsigned short* __restrict__ out_bf,
               float* __restrict__ out_f32,
               const unsigned short* __restrict__ hpre,
               const unsigned short* __restrict__ wbuf,
               const float* __restrict__ a0g,
               const float* __restrict__ a1g,
               int Ndim, int Kdim)
{
    __shared__ __align__(16) unsigned short As[2][TM][LDP];  // 40960 B
    __shared__ __align__(16) unsigned short Bs[2][TN][LDP];  // 20480 B

    const int tid  = threadIdx.x;
    const int m0   = blockIdx.y * TM;
    const int n0   = blockIdx.x * TN;
    const int lane = tid & 31;
    const int wid  = tid >> 5;
    const int wm   = wid & 3;        // 4 waves along M (64 rows each)
    const int wn   = wid >> 2;       // 2 waves along N (64 cols each)
    const int l16  = lane & 15;
    const int hi   = lane >> 4;

    v8f acc[4][4];
    #pragma unroll
    for (int mt = 0; mt < 4; ++mt)
        #pragma unroll
        for (int nt = 0; nt < 4; ++nt)
            acc[mt][nt] = (v8f){0.f,0.f,0.f,0.f,0.f,0.f,0.f,0.f};

    // A-loader: one thread per row, 32 k-elems
    const int a_row = tid;
    const int a_r   = m0 + a_row;
    const int sp_j  = a_r % 56;
    const int sp_i  = (a_r / 56) % 56;
    const int sp_bt = a_r / 3136;          // b*T + t
    const int sp_b  = a_r / 25088;
    // B-loader: 8 threads per k-row, 16 n-elems each
    const int b_noff = (tid & 7) * 16;
    const int b_krow = tid >> 3;

    auto loadA = [&](int kt2, int buf) {
        const int k0 = kt2 * TK;
        if (MODE == 0) {
            const float4* src = (const float4*)(Af32 + (size_t)a_r * Kdim + k0);
            unsigned* dst = (unsigned*)&As[buf][a_row][0];
            #pragma unroll
            for (int q = 0; q < 8; ++q) {
                float4 v = src[q];
                dst[q*2+0] = pack2bf(v.x, v.y);
                dst[q*2+1] = pack2bf(v.z, v.w);
            }
        } else if (MODE == 2) {
            const uint4* src = (const uint4*)(Abf + (size_t)a_r * Kdim + k0);
            uint4* dst = (uint4*)&As[buf][a_row][0];
            #pragma unroll
            for (int q = 0; q < 4; ++q) dst[q] = src[q];
        } else if (MODE == 1) {
            #pragma unroll
            for (int e = 0; e < 32; ++e) {
                int k  = k0 + e;
                int g  = k / 114;                      // 9 groups of 114 over HID=1024
                int si = sp_i - c_sh[g];
                int sj = sp_j - c_sw[g];
                unsigned short val = 0;
                if ((unsigned)si < 56u && (unsigned)sj < 56u)
                    val = Abf[((size_t)(sp_bt*3136 + si*56 + sj)) * 1024 + k];
                As[buf][a_row][e] = val;
            }
        } else { // MODE 3
            #pragma unroll
            for (int e = 0; e < 32; ++e) {
                int c  = k0 + e;
                int g  = c / 57;                       // 9 groups of 57 over C=512
                int si = sp_i + c_sh[g];
                int sj = sp_j + c_sw[g];
                float hv = 0.f;
                if ((unsigned)si < 56u && (unsigned)sj < 56u)
                    hv = bf2f(hpre[((size_t)(sp_bt*3136 + si*56 + sj)) * 512 + c]);
                float wv = bf2f(wbuf[(size_t)a_r * 512 + c]);
                float fv = a0g[sp_b * 512 + c] * hv + a1g[sp_b * 512 + c] * wv;
                As[buf][a_row][e] = f2bf(fv);
            }
        }
    };
    auto loadB = [&](int kt2, int buf) {
        const int k0 = kt2 * TK;
        const float4* src = (const float4*)(Bw + (size_t)(k0 + b_krow) * Ndim + n0 + b_noff);
        #pragma unroll
        for (int q = 0; q < 4; ++q) {
            float4 v = src[q];
            Bs[buf][b_noff + q*4 + 0][b_krow] = f2bf(v.x);
            Bs[buf][b_noff + q*4 + 1][b_krow] = f2bf(v.y);
            Bs[buf][b_noff + q*4 + 2][b_krow] = f2bf(v.z);
            Bs[buf][b_noff + q*4 + 3][b_krow] = f2bf(v.w);
        }
    };

    const int ksteps = Kdim / TK;
    loadA(0, 0);
    loadB(0, 0);
    __syncthreads();

    for (int kt = 0; kt < ksteps; ++kt) {
        const int cur = kt & 1;

        // prefetch K+2 tile into L2 (global_prefetch_b8)
        if (kt + 2 < ksteps) {
            if (MODE == 0)      __builtin_prefetch(Af32 + (size_t)a_r * Kdim + (kt + 2) * TK, 0, 3);
            else if (MODE == 2) __builtin_prefetch(Abf  + (size_t)a_r * Kdim + (kt + 2) * TK, 0, 3);
            __builtin_prefetch(Bw + (size_t)((kt + 2) * TK + b_krow) * Ndim + n0 + b_noff, 0, 3);
        }
        // issue next tile's global loads + LDS fills, then overlap with WMMA below
        if (kt + 1 < ksteps) {
            loadA(kt + 1, cur ^ 1);
            loadB(kt + 1, cur ^ 1);
        }

        // ---- fragments (two ds_load_b128 each) + WMMA ----
        AF afr[4];
        #pragma unroll
        for (int mt = 0; mt < 4; ++mt) {
            int row = wm * 64 + mt * 16 + l16;
            const uint4* ap = (const uint4*)&As[cur][row][hi * 8];
            afr[mt].q[0] = ap[0];          // kk = hi*8 + 0..7
            afr[mt].q[1] = ap[2];          // kk = 16 + hi*8 + 0..7
        }
        #pragma unroll
        for (int nt = 0; nt < 4; ++nt) {
            AF bfr;
            int ncol = wn * 64 + nt * 16 + l16;
            const uint4* bp = (const uint4*)&Bs[cur][ncol][hi * 8];
            bfr.q[0] = bp[0];
            bfr.q[1] = bp[2];
            #pragma unroll
            for (int mt = 0; mt < 4; ++mt) {
                acc[mt][nt] = __builtin_amdgcn_wmma_f32_16x16x32_bf16(
                    false, afr[mt].v, false, bfr.v,
                    (short)0, acc[mt][nt], false, false);
            }
        }
        __syncthreads();
    }

    // ---- epilogue: bias (+gelu) + store ----
    #pragma unroll
    for (int mt = 0; mt < 4; ++mt)
        #pragma unroll
        for (int nt = 0; nt < 4; ++nt)
            #pragma unroll
            for (int rr = 0; rr < 8; ++rr) {
                int grow = m0 + wm * 64 + mt * 16 + rr + 8 * hi;
                int gcol = n0 + wn * 64 + nt * 16 + l16;
                float v = acc[mt][nt][rr] + bias[gcol];
                if (MODE != 3) {
                    v = gelu_f(v);
                    out_bf[(size_t)grow * Ndim + gcol] = f2bf(v);
                } else {
                    out_f32[(size_t)grow * Ndim + gcol] = v;
                }
            }
}

// Deterministic partial reduction of (h_gathered + w) over 128-row chunks.
__global__ __launch_bounds__(256)
void reduce_hw(const unsigned short* __restrict__ hpre,
               const unsigned short* __restrict__ wbuf,
               float* __restrict__ partials)
{
    int chunk = blockIdx.x;          // 0..391, chunks 0..195 = batch 0
    int r0 = chunk * 128;
    int tid = threadIdx.x;
    for (int cc = 0; cc < 2; ++cc) {
        int c = tid + cc * 256;
        int g = c / 57;
        int dsh = c_sh[g], dsw = c_sw[g];
        float s = 0.f;
        for (int rr = 0; rr < 128; ++rr) {
            int r  = r0 + rr;
            int j  = r % 56, i = (r / 56) % 56, bt = r / 3136;
            int si = i + dsh, sj = j + dsw;
            float hv = 0.f;
            if ((unsigned)si < 56u && (unsigned)sj < 56u)
                hv = bf2f(hpre[((size_t)(bt*3136 + si*56 + sj)) * 512 + c]);
            s += hv + bf2f(wbuf[(size_t)r * 512 + c]);
        }
        partials[(size_t)chunk * 512 + c] = s;
    }
}

// Single block: final reduce (fixed order), tiny MLP gate, pairwise softmax.
__global__ __launch_bounds__(256)
void gate_kernel(const float* __restrict__ partials,
                 const float* __restrict__ rw1_w, const float* __restrict__ rw1_b,
                 const float* __restrict__ rw2_w, const float* __restrict__ rw2_b,
                 float* __restrict__ a0g, float* __restrict__ a1g)
{
    __shared__ float mean_s[2][512];
    __shared__ float hid_s[2][128];
    __shared__ float av_s[2][1024];
    int tid = threadIdx.x;

    for (int idx = tid; idx < 1024; idx += 256) {
        int b = idx >> 9, c = idx & 511;
        float s = 0.f;
        for (int ch = 0; ch < 196; ++ch)
            s += partials[(size_t)(b * 196 + ch) * 512 + c];
        mean_s[b][c] = s * (1.0f / 25088.0f);
    }
    __syncthreads();
    for (int idx = tid; idx < 256; idx += 256) {
        int b = idx >> 7, o = idx & 127;
        float acc = rw1_b[o];
        for (int c = 0; c < 512; ++c) acc += mean_s[b][c] * rw1_w[c * 128 + o];
        hid_s[b][o] = gelu_f(acc);
    }
    __syncthreads();
    for (int idx = tid; idx < 2048; idx += 256) {
        int b = idx >> 10, o = idx & 1023;
        float acc = rw2_b[o];
        for (int c = 0; c < 128; ++c) acc += hid_s[b][c] * rw2_w[c * 1024 + o];
        av_s[b][o] = acc;
    }
    __syncthreads();
    for (int idx = tid; idx < 1024; idx += 256) {
        int b = idx >> 9, c = idx & 511;
        float v0 = av_s[b][2 * c], v1 = av_s[b][2 * c + 1];
        float mx = fmaxf(v0, v1);
        float e0 = expf(v0 - mx), e1 = expf(v1 - mx);
        float inv = 1.0f / (e0 + e1);
        a0g[b * 512 + c] = e0 * inv;
        a1g[b * 512 + c] = e1 * inv;
    }
}

extern "C" void kernel_launch(void* const* d_in, const int* in_sizes, int n_in,
                              void* d_out, int out_size, void* d_ws, size_t ws_size,
                              hipStream_t stream) {
    const float* x      = (const float*)d_in[0];
    const float* fc_w   = (const float*)d_in[1];
    const float* fc_b   = (const float*)d_in[2];
    const float* fc1_w  = (const float*)d_in[3];
    const float* fc1_b  = (const float*)d_in[4];
    const float* fc2_w  = (const float*)d_in[5];
    const float* fc2_b  = (const float*)d_in[6];
    const float* rw1_w  = (const float*)d_in[7];
    const float* rw1_b  = (const float*)d_in[8];
    const float* rw2_w  = (const float*)d_in[9];
    const float* rw2_b  = (const float*)d_in[10];
    const float* proj_w = (const float*)d_in[11];
    const float* proj_b = (const float*)d_in[12];

    char* ws = (char*)d_ws;
    const size_t XH_B = (size_t)50176 * 1024 * 2;   // bf16 xh
    const size_t HW_B = (size_t)50176 * 512 * 2;    // bf16 hpre / wbuf
    unsigned short* xh   = (unsigned short*)ws;
    unsigned short* hpre = (unsigned short*)(ws + XH_B);
    unsigned short* wbuf = (unsigned short*)(ws + XH_B + HW_B);
    float* partials = (float*)(ws + XH_B + 2 * HW_B);
    float* a0g = partials + (size_t)392 * 512;
    float* a1g = a0g + 1024;

    dim3 blk(256);
    // K1: xh = gelu(x @ fc_w + fc_b)           [50176,512]x[512,1024]
    gemm_tile<0><<<dim3(1024 / TN, 50176 / TM), blk, 0, stream>>>(
        x, nullptr, fc_w, fc_b, xh, nullptr, nullptr, nullptr, nullptr, nullptr, 1024, 512);
    // K2a: hpre = gelu(shift(xh) @ fc1_w + b)  [50176,1024]x[1024,512]
    gemm_tile<1><<<dim3(512 / TN, 50176 / TM), blk, 0, stream>>>(
        nullptr, xh, fc1_w, fc1_b, hpre, nullptr, nullptr, nullptr, nullptr, nullptr, 512, 1024);
    // K2b: wbuf = gelu(xh @ fc2_w + b)
    gemm_tile<2><<<dim3(512 / TN, 50176 / TM), blk, 0, stream>>>(
        nullptr, xh, fc2_w, fc2_b, wbuf, nullptr, nullptr, nullptr, nullptr, nullptr, 512, 1024);
    // K3: deterministic partial sums of (h + w)
    reduce_hw<<<dim3(392), blk, 0, stream>>>(hpre, wbuf, partials);
    // K4: gating MLP + softmax
    gate_kernel<<<dim3(1), blk, 0, stream>>>(partials, rw1_w, rw1_b, rw2_w, rw2_b, a0g, a1g);
    // K5: out = (a0*h + a1*w) @ proj_w + proj_b (fuses inverse shift + scaling into A loader)
    gemm_tile<3><<<dim3(512 / TN, 50176 / TM), blk, 0, stream>>>(
        nullptr, nullptr, proj_w, proj_b, nullptr, (float*)d_out, hpre, wbuf, a0g, a1g, 512, 512);
}
